// ResRnn_19284403159333
// MI455X (gfx1250) — matrix operand order repaired and checked by reading, not compiled, not run on hardware
//
#include <hip/hip_runtime.h>
#include <cstdint>
#include <cstddef>

// ---------------------------------------------------------------------------
// ResRnn on MI455X (gfx1250, wave32, WMMA).
//   s_t = 0.9 * concat(x_t, s_{t-1}[:, :1984]) + 0.1 * (|A W1^T + b1| W2^T + b2)
// bf16 operands + f32 accumulation via v_wmma_f32_16x16x32_bf16.
// Whole working set (~22 MB) is L2-resident (192 MB): 256 dependent GEMMs fed
// from L2/LDS, never HBM.  W tiles staged with the CDNA5 async global->LDS
// path (ASYNCcnt) when available.
// ---------------------------------------------------------------------------

typedef __attribute__((ext_vector_type(16))) __bf16 v16bf;
typedef __attribute__((ext_vector_type(8)))  float  v8f;
typedef int v4i_vs __attribute__((__vector_size__(16)));

#define WIDTH_  2048
#define STATE_  1984
#define INPUT_  64
#define BATCH_  128
#define SEQ_    128
#define ID_P    0.9f

#define GLOBAL_AS __attribute__((address_space(1)))
#define LDS_AS    __attribute__((address_space(3)))

#if __has_builtin(__builtin_amdgcn_global_load_async_to_lds_b128)
#define HAVE_ASYNC_LDS 1
#else
#define HAVE_ASYNC_LDS 0
#endif

union Frag16 { v16bf v; uint4 q[2]; };

__device__ __forceinline__ uint4 ldg16(const void* p) {
  return *reinterpret_cast<const uint4*>(p);
}

__device__ __forceinline__ void stage_w16(const __bf16* gp, __bf16* lp) {
#if HAVE_ASYNC_LDS
  // Direct global->LDS DMA, no VGPR round trip; tracked by ASYNCcnt.
  // Param 0: AS1 (global) v4i*, param 1: AS3 (LDS) v4i*, then imm offset, cpol.
  __builtin_amdgcn_global_load_async_to_lds_b128(
      (GLOBAL_AS v4i_vs*)gp, (LDS_AS v4i_vs*)lp, /*offset=*/0, /*cpol=*/0);
#else
  *reinterpret_cast<uint4*>(lp) = ldg16(gp);
#endif
}

__device__ __forceinline__ void wait_async_then_barrier() {
#if HAVE_ASYNC_LDS
#if __has_builtin(__builtin_amdgcn_s_wait_asynccnt)
  __builtin_amdgcn_s_wait_asynccnt(0);
#else
  asm volatile("s_wait_asynccnt 0x0" ::: "memory");
#endif
#endif
  __syncthreads();
}

// ---------------------------------------------------------------------------
// Fused GEMM step.  C = A(128x2048, bf16) * W^T(2048x2048, bf16) + bias.
// MODE 1: A = shifted view of [Xbf_t | Sbf_prev];  out_bf = |C| as bf16 (H).
// MODE 2: A = Hbf; out = 0.9*resid + 0.1*C written as fp32 state + bf16 mirror.
//
// Grid: 32 blocks x 256 threads (8 waves). Block owns M=128 x N=64.
// Wave w owns rows 16w..16w+15 and all 4 N-tiles (4 f32 accumulators).
// W tile (64 rows x 32 k, 4 KB) is double-buffered in LDS, shared by 8 waves.
// ---------------------------------------------------------------------------
template <int MODE>
__global__ __launch_bounds__(256) void gemm_step(
    const __bf16* __restrict__ Abf,      // MODE1: Sbf_prev ; MODE2: Hbf
    const __bf16* __restrict__ Xbf_t,    // MODE1 only
    const __bf16* __restrict__ W,        // [2048][2048] bf16 row-major
    const float*  __restrict__ bias,
    const float*  __restrict__ Sfp_prev, // MODE2 residual
    const float*  __restrict__ Xfp_t,    // MODE2 residual
    __bf16* __restrict__ out_bf,         // MODE1: Hbf ; MODE2: Sbf_new
    float*  __restrict__ out_fp)         // MODE2: Sfp_new
{
  __shared__ __align__(16) __bf16 ldsB[2][64 * 32];  // 8 KB double buffer

  const int tid   = threadIdx.x;
  const int lane  = tid & 31;
  const int wave  = tid >> 5;            // 0..7 -> M tile
  const int nbase = blockIdx.x * 64;

  // A fragment addressing (ISA 16-bit A 16x32 layout):
  //   lane L: row M = L&15, chunks K = khalf..khalf+7 and 16+khalf..16+khalf+7
  const int mA    = wave * 16 + (lane & 15);
  const int khalf = (lane >> 4) * 8;

  // W staging: thread i moves one 16B chunk: row = i>>2 (0..63), kcol = (i&3)*8
  const int brow = tid >> 2;
  const int bcol = (tid & 3) * 8;
  const __bf16* wrow = W + (size_t)(nbase + brow) * WIDTH_;

  v8f acc[4] = {};

  // prologue: stage K-chunk 0 into LDS buffer 0
  stage_w16(wrow + bcol, &ldsB[0][brow * 32 + bcol]);
  wait_async_then_barrier();

  for (int it = 0; it < 64; ++it) {
    const int buf = it & 1;
    const int kc  = it * 32;

    if (it + 1 < 64) {  // stage next W tile while computing current
      stage_w16(wrow + kc + 32 + bcol, &ldsB[buf ^ 1][brow * 32 + bcol]);
    }
    if (it + 2 < 64) {  // near-scope prefetch of the tile after that
      __builtin_prefetch(wrow + kc + 64 + bcol, 0, 3);
    }

    // A fragment: two 16B contiguous chunks per lane.
    Frag16 a;
    const int ks0 = kc + khalf;
    const int ks1 = kc + 16 + khalf;
    if (MODE == 1) {
      // shifted concat view; chunks never straddle the k=64 boundary
      a.q[0] = (ks0 < INPUT_)
                   ? ldg16(Xbf_t + mA * INPUT_ + ks0)
                   : ldg16(Abf + (size_t)mA * WIDTH_ + (ks0 - INPUT_));
      a.q[1] = (ks1 < INPUT_)
                   ? ldg16(Xbf_t + mA * INPUT_ + ks1)
                   : ldg16(Abf + (size_t)mA * WIDTH_ + (ks1 - INPUT_));
    } else {
      a.q[0] = ldg16(Abf + (size_t)mA * WIDTH_ + ks0);
      a.q[1] = ldg16(Abf + (size_t)mA * WIDTH_ + ks1);
    }

    // Load ALL four B fragments first (independent registers), then issue the
    // 4-WMMA burst -- avoids dscnt waits between consecutive v_wmma ops.
    const int bk = (lane >> 4) * 8;
    Frag16 bfr[4];
#pragma unroll
    for (int nt = 0; nt < 4; ++nt) {
      const int row = nt * 16 + (lane & 15);
      bfr[nt].q[0] = *reinterpret_cast<const uint4*>(&ldsB[buf][row * 32 + bk]);
      bfr[nt].q[1] =
          *reinterpret_cast<const uint4*>(&ldsB[buf][row * 32 + bk + 16]);
    }
#pragma unroll
    for (int nt = 0; nt < 4; ++nt) {
      acc[nt] = __builtin_amdgcn_wmma_f32_16x16x32_bf16(
          false, a.v, false, bfr[nt].v, (short)0, acc[nt], false, false);
    }
    wait_async_then_barrier();
  }

  // Epilogue. D layout: lanes 0-15: N=lane, M=r (VGPR r); lanes 16-31: M=8+r.
  const int mrow0 = wave * 16 + (lane >> 4) * 8;
  const int ncol  = lane & 15;
#pragma unroll
  for (int nt = 0; nt < 4; ++nt) {
    const int n    = nbase + nt * 16 + ncol;
    const float bn = bias[n];
#pragma unroll
    for (int r = 0; r < 8; ++r) {
      const int   mm = mrow0 + r;
      const float g  = acc[nt][r] + bn;
      if (MODE == 1) {
        out_bf[(size_t)mm * WIDTH_ + n] = (__bf16)fabsf(g);
      } else {
        const float resid = (n < INPUT_)
                                ? Xfp_t[mm * INPUT_ + n]
                                : Sfp_prev[(size_t)mm * WIDTH_ + (n - INPUT_)];
        const float v = resid * ID_P + (1.0f - ID_P) * g;
        out_fp[(size_t)mm * WIDTH_ + n] = v;
        out_bf[(size_t)mm * WIDTH_ + n] = (__bf16)v;
      }
    }
  }
}

// fp32 -> bf16, 4 elements per thread (all sizes divisible by 1024)
__global__ __launch_bounds__(256) void cvt_f32_bf16_x4(
    const float* __restrict__ in, __bf16* __restrict__ out, int n4) {
  const int i = blockIdx.x * 256 + threadIdx.x;
  if (i < n4) {
    const float4 f = reinterpret_cast<const float4*>(in)[i];
    out[4 * i + 0] = (__bf16)f.x;
    out[4 * i + 1] = (__bf16)f.y;
    out[4 * i + 2] = (__bf16)f.z;
    out[4 * i + 3] = (__bf16)f.w;
  }
}

// zero-init state (harness poisons workspace with 0xAA)
__global__ __launch_bounds__(256) void zero_state(
    float* __restrict__ sfp, __bf16* __restrict__ sbf) {
  const int i = blockIdx.x * 256 + threadIdx.x;  // 262144 total
  sfp[i] = 0.0f;
  sbf[i] = (__bf16)0.0f;
}

// out[b, o] = S[b, 1984 + o]
__global__ __launch_bounds__(256) void extract_out(
    const float* __restrict__ S, float* __restrict__ out) {
  const int i = blockIdx.x * 256 + threadIdx.x;  // 8192 total
  const int b = i >> 6;
  const int o = i & 63;
  out[i] = S[(size_t)b * WIDTH_ + STATE_ + o];
}

extern "C" void kernel_launch(void* const* d_in, const int* in_sizes, int n_in,
                              void* d_out, int out_size, void* d_ws,
                              size_t ws_size, hipStream_t stream) {
  const float* X  = (const float*)d_in[0];  // [128,128,64]
  const float* W1 = (const float*)d_in[1];  // [2048,2048]
  const float* b1 = (const float*)d_in[2];  // [2048]
  const float* W2 = (const float*)d_in[3];  // [2048,2048]
  const float* b2 = (const float*)d_in[4];  // [2048]

  // workspace layout (~21.5 MiB, 16B-aligned slabs)
  char* ws = (char*)d_ws;
  __bf16* W1bf   = (__bf16*)(ws + 0);          //  8 MiB
  __bf16* W2bf   = (__bf16*)(ws + 8388608);    //  8 MiB
  __bf16* Xbf    = (__bf16*)(ws + 16777216);   //  2 MiB
  __bf16* Hbf    = (__bf16*)(ws + 18874368);   //  512 KiB
  __bf16* Sbf[2] = {(__bf16*)(ws + 19398656), (__bf16*)(ws + 19922944)};
  float*  Sfp[2] = {(float*)(ws + 20447232), (float*)(ws + 21495808)};

  // one-time precision conversion of weights + inputs into L2-resident bf16
  cvt_f32_bf16_x4<<<4096, 256, 0, stream>>>(W1, W1bf, 1048576);
  cvt_f32_bf16_x4<<<4096, 256, 0, stream>>>(W2, W2bf, 1048576);
  cvt_f32_bf16_x4<<<1024, 256, 0, stream>>>(X, Xbf, 262144);
  zero_state<<<1024, 256, 0, stream>>>(Sfp[0], Sbf[0]);

  for (int t = 0; t < SEQ_; ++t) {
    const int p = t & 1;
    const __bf16* Xbf_t = Xbf + (size_t)t * BATCH_ * INPUT_;
    const float*  Xfp_t = X   + (size_t)t * BATCH_ * INPUT_;
    // H = |concat(x_t, s[:, :1984]) @ W1^T + b1|
    gemm_step<1><<<32, 256, 0, stream>>>(Sbf[p], Xbf_t, W1bf, b1,
                                         nullptr, nullptr, Hbf, nullptr);
    // s' = 0.9*concat(...) + 0.1*(H @ W2^T + b2)
    gemm_step<2><<<32, 256, 0, stream>>>(Hbf, nullptr, W2bf, b2,
                                         Sfp[p], Xfp_t, Sbf[p ^ 1], Sfp[p ^ 1]);
  }
  // after t=127 (p=1) the final state lives in buffer 0
  extract_out<<<32, 256, 0, stream>>>(Sfp[0], (float*)d_out);
}
